// edge_node_block_78151224918195
// MI455X (gfx1250) — compile-verified, matrix operand order checked by reading, not run on hardware
//
#include <hip/hip_runtime.h>
#include <hip/hip_bf16.h>

typedef __attribute__((ext_vector_type(16))) _Float16 v16h;
typedef __attribute__((ext_vector_type(8)))  float    v8f;

union Frag16 { v16h v; unsigned u[8]; };

__device__ __forceinline__ float silu_f(float x) {
    return x / (1.0f + __expf(-x));
}

// CDNA5 16-bit A/B fragment K-offset for VGPR pair v; lanes split K by 8.
__device__ __forceinline__ int kmap(int v) { return 16 * (v >> 2) + 2 * (v & 3); }

__global__ void zero_kernel(float* p, int n) {
    int i = (blockIdx.x * blockDim.x + threadIdx.x) * 4;
    if (i < n) { float4 z = {0.f, 0.f, 0.f, 0.f}; *(float4*)(p + i) = z; }
}

// f32 -> f16 pre-cast of node features (so edge gathers move half the bytes
// and the hot loop has no per-edge cvt work).
__global__ void cast_kernel(const float* __restrict__ in, _Float16* __restrict__ out, int n) {
    int i = (blockIdx.x * blockDim.x + threadIdx.x) * 4;
    if (i < n) {
        float4 v = *(const float4*)(in + i);
        union { _Float16 h[4]; unsigned long long q; } u;
        u.h[0] = (_Float16)v.x; u.h[1] = (_Float16)v.y;
        u.h[2] = (_Float16)v.z; u.h[3] = (_Float16)v.w;
        *(unsigned long long*)(out + i) = u.q;
    }
}

// ---------------------------------------------------------------------------
// Edge message kernel: per wave32, one tile of 16 edges.
//   f[16 x 160] = [h_src(64) | h_dst(64) | radial(1) | a(16) | pad(15)]
//   m   = SiLU(f @ W_e1 + b_e1)        (20 wmma: 5 K-chunks x 4 N-tiles)
//   msg = SiLU(m @ W_e2 + b_e2)        ( 8 wmma: 2 K-chunks x 4 N-tiles)
//   atomicAdd into h_neigh[dst]
// ---------------------------------------------------------------------------
__global__ void __launch_bounds__(128)
edge_kernel(const _Float16* __restrict__ node_h, const float* __restrict__ coord,
            const float* __restrict__ edge_feat, const int* __restrict__ src,
            const int* __restrict__ dst,
            const float* __restrict__ We1, const float* __restrict__ be1,
            const float* __restrict__ We2, const float* __restrict__ be2,
            float* __restrict__ h_neigh, int ntiles)
{
    __shared__ _Float16 Wt1[64 * 160];     // W_e1 transposed [h][k], k padded to 160
    __shared__ _Float16 Wt2[64 * 64];      // W_e2 transposed [h][k]
    __shared__ _Float16 fstg[4][16 * 160]; // per-wave staging (reused as 16x64 for m)
    __shared__ int      didx[4][16];
    __shared__ int      sidx[4][16];

    const int tid  = threadIdx.x;
    const int wave = tid >> 5;
    const int lane = tid & 31;

    // Convert+transpose weights into LDS (cooperative, once per block)
    for (int i = tid; i < 145 * 64; i += blockDim.x) {
        int k = i / 64, h = i % 64;
        Wt1[h * 160 + k] = (_Float16)We1[i];
    }
    for (int i = tid; i < 64 * 15; i += blockDim.x) {     // zero pad k = 145..159
        int h = i / 15, k = 145 + i % 15;
        Wt1[h * 160 + k] = (_Float16)0.f;
    }
    for (int i = tid; i < 64 * 64; i += blockDim.x) {
        int k = i / 64, h = i % 64;
        Wt2[h * 64 + k] = (_Float16)We2[i];
    }
    __syncthreads();

    const int m   = lane & 15;  // A-layout row / C-layout column index
    const int kHi = lane >> 4;

    _Float16* F = fstg[wave];
    int* S = sidx[wave];
    int* D = didx[wave];

    for (int tile = blockIdx.x * 4 + wave; tile < ntiles; tile += gridDim.x * 4) {
        const int base = tile << 4;

        // ---- stage indices, radial, zero-pad ----
        if (lane < 16) {
            int e = base + lane;
            int s = src[e], d = dst[e];
            S[lane] = s; D[lane] = d;
            float dx = coord[s * 3 + 0] - coord[d * 3 + 0];
            float dy = coord[s * 3 + 1] - coord[d * 3 + 1];
            float dz = coord[s * 3 + 2] - coord[d * 3 + 2];
            F[lane * 160 + 128] = (_Float16)(dx * dx + dy * dy + dz * dz);
        } else {
            int mm = lane - 16;
            for (int k = 145; k < 160; ++k) F[mm * 160 + k] = (_Float16)0.f;
        }
        asm volatile("s_wait_dscnt 0" ::: "memory");

        // ---- stage features: f16 row gathers, 2 halves per lane per row ----
        for (int mm = 0; mm < 16; ++mm) {
            int s = S[mm], d = D[mm];
            *(unsigned*)&F[mm * 160 + 2 * lane] =
                *(const unsigned*)&node_h[s * 64 + 2 * lane];
            *(unsigned*)&F[mm * 160 + 64 + 2 * lane] =
                *(const unsigned*)&node_h[d * 64 + 2 * lane];
            if (lane < 16)
                F[mm * 160 + 129 + lane] = (_Float16)edge_feat[(base + mm) * 16 + lane];
        }
        asm volatile("s_wait_dscnt 0" ::: "memory");

        // ---- build A fragments (16x32 f16 each, 5 K-chunks) ----
        Frag16 A[5];
        const _Float16* rowp = &F[m * 160];
#pragma unroll
        for (int c = 0; c < 5; ++c) {
            int kb = c * 32 + 8 * kHi;
#pragma unroll
            for (int v = 0; v < 8; ++v)
                A[c].u[v] = *(const unsigned*)(rowp + kb + kmap(v));
        }

        // ---- GEMM1: [16,160] x [160,64], bias + SiLU, re-stage as 16x64 f16 ----
#pragma unroll
        for (int t = 0; t < 4; ++t) {
            int col = t * 16 + (lane & 15);      // C layout: lane fixes output col
            v8f acc = {};
#pragma unroll
            for (int c = 0; c < 5; ++c) {
                Frag16 B;
                const _Float16* wp = &Wt1[col * 160 + c * 32 + 8 * kHi];
#pragma unroll
                for (int v = 0; v < 8; ++v)
                    B.u[v] = *(const unsigned*)(wp + kmap(v));
                acc = __builtin_amdgcn_wmma_f32_16x16x32_f16(
                    false, A[c].v, false, B.v, (short)0, acc, false, false);
            }
            float bias = be1[col];
            int mHi = lane >> 4;
#pragma unroll
            for (int v = 0; v < 8; ++v) {
                float y = silu_f(acc[v] + bias);
                F[(8 * mHi + v) * 64 + col] = (_Float16)y;   // reuse staging as m-matrix
            }
        }
        asm volatile("s_wait_dscnt 0" ::: "memory");

        // ---- GEMM2: [16,64] x [64,64], bias + SiLU, scatter-add ----
        Frag16 A2[2];
        const _Float16* rowp2 = &F[m * 64];
#pragma unroll
        for (int c = 0; c < 2; ++c) {
            int kb = c * 32 + 8 * kHi;
#pragma unroll
            for (int v = 0; v < 8; ++v)
                A2[c].u[v] = *(const unsigned*)(rowp2 + kb + kmap(v));
        }
#pragma unroll
        for (int t = 0; t < 4; ++t) {
            int col = t * 16 + (lane & 15);
            v8f acc = {};
#pragma unroll
            for (int c = 0; c < 2; ++c) {
                Frag16 B;
                const _Float16* wp = &Wt2[col * 64 + c * 32 + 8 * kHi];
#pragma unroll
                for (int v = 0; v < 8; ++v)
                    B.u[v] = *(const unsigned*)(wp + kmap(v));
                acc = __builtin_amdgcn_wmma_f32_16x16x32_f16(
                    false, A2[c].v, false, B.v, (short)0, acc, false, false);
            }
            float bias = be2[col];
            int mHi = lane >> 4;
#pragma unroll
            for (int v = 0; v < 8; ++v) {
                float y = silu_f(acc[v] + bias);
                atomicAdd(&h_neigh[D[8 * mHi + v] * 64 + col], y);
            }
        }
    }
}

// ---------------------------------------------------------------------------
// Node update kernel: per wave32, one tile of 16 nodes.
//   z = [node_feat(64) | h_neigh(64)]   (K = 128)
//   h = SiLU(z @ W_n1 + b_n1) @ W_n2 + b_n2
// ---------------------------------------------------------------------------
__global__ void __launch_bounds__(128)
node_kernel(const _Float16* __restrict__ node_h, const float* __restrict__ h_neigh,
            const float* __restrict__ Wn1, const float* __restrict__ bn1,
            const float* __restrict__ Wn2, const float* __restrict__ bn2,
            float* __restrict__ out, int ntiles)
{
    __shared__ _Float16 Wt3[64 * 128];     // W_n1 transposed [h][k]
    __shared__ _Float16 Wt4[64 * 64];      // W_n2 transposed [h][k]
    __shared__ _Float16 zstg[4][16 * 128]; // per-wave staging (reused 16x64 for m)

    const int tid  = threadIdx.x;
    const int wave = tid >> 5;
    const int lane = tid & 31;

    for (int i = tid; i < 128 * 64; i += blockDim.x) {
        int k = i / 64, h = i % 64;
        Wt3[h * 128 + k] = (_Float16)Wn1[i];
    }
    for (int i = tid; i < 64 * 64; i += blockDim.x) {
        int k = i / 64, h = i % 64;
        Wt4[h * 64 + k] = (_Float16)Wn2[i];
    }
    __syncthreads();

    const int m   = lane & 15;
    const int kHi = lane >> 4;
    _Float16* Z = zstg[wave];

    for (int tile = blockIdx.x * 4 + wave; tile < ntiles; tile += gridDim.x * 4) {
        const int base = tile << 4;

        for (int mm = 0; mm < 16; ++mm) {
            int n = base + mm;
            *(unsigned*)&Z[mm * 128 + 2 * lane] =
                *(const unsigned*)&node_h[n * 64 + 2 * lane];
            Z[mm * 128 + 64 + lane] = (_Float16)h_neigh[n * 64 + lane];
            Z[mm * 128 + 96 + lane] = (_Float16)h_neigh[n * 64 + 32 + lane];
        }
        asm volatile("s_wait_dscnt 0" ::: "memory");

        Frag16 A[4];
        const _Float16* rowp = &Z[m * 128];
#pragma unroll
        for (int c = 0; c < 4; ++c) {
            int kb = c * 32 + 8 * kHi;
#pragma unroll
            for (int v = 0; v < 8; ++v)
                A[c].u[v] = *(const unsigned*)(rowp + kb + kmap(v));
        }

#pragma unroll
        for (int t = 0; t < 4; ++t) {
            int col = t * 16 + (lane & 15);
            v8f acc = {};
#pragma unroll
            for (int c = 0; c < 4; ++c) {
                Frag16 B;
                const _Float16* wp = &Wt3[col * 128 + c * 32 + 8 * kHi];
#pragma unroll
                for (int v = 0; v < 8; ++v)
                    B.u[v] = *(const unsigned*)(wp + kmap(v));
                acc = __builtin_amdgcn_wmma_f32_16x16x32_f16(
                    false, A[c].v, false, B.v, (short)0, acc, false, false);
            }
            float bias = bn1[col];
            int mHi = lane >> 4;
#pragma unroll
            for (int v = 0; v < 8; ++v) {
                float y = silu_f(acc[v] + bias);
                Z[(8 * mHi + v) * 64 + col] = (_Float16)y;
            }
        }
        asm volatile("s_wait_dscnt 0" ::: "memory");

        Frag16 A2[2];
        const _Float16* rowp2 = &Z[m * 64];
#pragma unroll
        for (int c = 0; c < 2; ++c) {
            int kb = c * 32 + 8 * kHi;
#pragma unroll
            for (int v = 0; v < 8; ++v)
                A2[c].u[v] = *(const unsigned*)(rowp2 + kb + kmap(v));
        }
#pragma unroll
        for (int t = 0; t < 4; ++t) {
            int col = t * 16 + (lane & 15);
            v8f acc = {};
#pragma unroll
            for (int c = 0; c < 2; ++c) {
                Frag16 B;
                const _Float16* wp = &Wt4[col * 64 + c * 32 + 8 * kHi];
#pragma unroll
                for (int v = 0; v < 8; ++v)
                    B.u[v] = *(const unsigned*)(wp + kmap(v));
                acc = __builtin_amdgcn_wmma_f32_16x16x32_f16(
                    false, A2[c].v, false, B.v, (short)0, acc, false, false);
            }
            float bias = bn2[col];
            int mHi = lane >> 4;
#pragma unroll
            for (int v = 0; v < 8; ++v) {
                float y = acc[v] + bias;                       // no activation on layer 2
                out[(base + 8 * mHi + v) * 64 + col] = y;
            }
        }
    }
}

extern "C" void kernel_launch(void* const* d_in, const int* in_sizes, int n_in,
                              void* d_out, int out_size, void* d_ws, size_t ws_size,
                              hipStream_t stream) {
    const float* node_feat = (const float*)d_in[0];
    const float* coord     = (const float*)d_in[1];
    const float* edge_feat = (const float*)d_in[2];
    const int*   src       = (const int*)d_in[3];
    const int*   dst       = (const int*)d_in[4];
    const float* We1 = (const float*)d_in[5];
    const float* be1 = (const float*)d_in[6];
    const float* We2 = (const float*)d_in[7];
    const float* be2 = (const float*)d_in[8];
    const float* Wn1 = (const float*)d_in[9];
    const float* bn1 = (const float*)d_in[10];
    const float* Wn2 = (const float*)d_in[11];
    const float* bn2 = (const float*)d_in[12];
    float* out = (float*)d_out;

    const int E = in_sizes[3];           // edge count
    const int N = in_sizes[0] / 64;      // node count
    const int acc_elems = N * 64;

    // workspace: [ h_neigh: N*64 f32 | node_h: N*64 f16 ]
    float*    h_neigh = (float*)d_ws;
    _Float16* node_h  = (_Float16*)((char*)d_ws + (size_t)acc_elems * sizeof(float));

    zero_kernel<<<(acc_elems / 4 + 255) / 256, 256, 0, stream>>>(h_neigh, acc_elems);
    cast_kernel<<<(acc_elems / 4 + 255) / 256, 256, 0, stream>>>(node_feat, node_h, acc_elems);

    const int etiles = E / 16;           // E = 800000 -> 50000 tiles
    edge_kernel<<<1250, 128, 0, stream>>>(node_h, coord, edge_feat, src, dst,
                                          We1, be1, We2, be2, h_neigh, etiles);

    const int ntiles = N / 16;           // N = 50000 -> 3125 tiles
    node_kernel<<<800, 128, 0, stream>>>(node_h, h_neigh, Wn1, bn1, Wn2, bn2,
                                         out, ntiles);
}
